// Encoder_79843442033106
// MI455X (gfx1250) — matrix operand order repaired and checked by reading, not compiled, not run on hardware
//
#include <hip/hip_runtime.h>
#include <hip/hip_fp16.h>

typedef _Float16 half_t;
typedef __attribute__((ext_vector_type(16))) _Float16 v16h;
typedef __attribute__((ext_vector_type(8)))  _Float16 v8h;
typedef __attribute__((ext_vector_type(8)))  float    v8f;

#define KD 2016            // 2*126*8 feature dim per query/codebook row
#define NSPLIT 8           // m-range splits (grid.y)
#define WAVES 8            // waves per block (256 threads, wave32)
#define NSLOTS (NSPLIT * WAVES)
#define BIG 3.0e38f

// ---------------------------------------------------------------------------
// Kernel 1: per codebook row, c2[m] = sum(c^2), cn2[m] = sum((1-c)^2)
// ---------------------------------------------------------------------------
__global__ void rowstats_kernel(const half_t* __restrict__ cb,
                                float* __restrict__ c2,
                                float* __restrict__ cn2,
                                int Mrows) {
    const int m = blockIdx.x;
    if (m >= Mrows) return;
    __shared__ float s1[256];
    __shared__ float s2[256];
    const half_t* row = cb + (size_t)m * KD;
    float a1 = 0.f, a2 = 0.f;
    for (int k = threadIdx.x; k < KD; k += 256) {
        float v = (float)row[k];
        a1 += v;
        a2 += v * v;
    }
    s1[threadIdx.x] = a1;
    s2[threadIdx.x] = a2;
    __syncthreads();
    for (int off = 128; off > 0; off >>= 1) {
        if (threadIdx.x < off) {
            s1[threadIdx.x] += s1[threadIdx.x + off];
            s2[threadIdx.x] += s2[threadIdx.x + off];
        }
        __syncthreads();
    }
    if (threadIdx.x == 0) {
        c2[m]  = s2[0];
        cn2[m] = (float)KD - 2.0f * s1[0] + s2[0];
    }
}

// ---------------------------------------------------------------------------
// Kernel 2: fused WMMA GEMM + distance + partial argmin.
// grid = (B, NSPLIT), block = 256 (8 wave32 waves).
// Each block owns one 16-query tile; each wave processes 32-wide m-tiles
// (two 16x16 WMMA tiles sharing one A fragment -> 2 WMMAs per LDS A read).
// ---------------------------------------------------------------------------
__global__ void __launch_bounds__(256)
encode_kernel(const half_t* __restrict__ x,    // [B,2,126,128]
              const half_t* __restrict__ cb,   // [M,2016]
              const float*  __restrict__ c2,
              const float*  __restrict__ cn2,
              float* __restrict__ pvals,       // [Q_total, NSLOTS]
              int*   __restrict__ pidx,        // [Q_total, NSLOTS]
              int Mrows) {
    __shared__ half_t ldsA[16 * KD];   // 64512 B: 16 queries x 2016 (row-major)
    __shared__ float  ldsX2[16];
    __shared__ float  ldsXS[16];

    const int tid  = threadIdx.x;
    const int wave = tid >> 5;
    const int lane = tid & 31;
    const int lo   = lane & 15;
    const int hi   = lane >> 4;

    if (tid < 16) { ldsX2[tid] = 0.f; ldsXS[tid] = 0.f; }
    __syncthreads();

    // ---- gather 16 queries into LDS (slot-of-8 layout -> packed rows),
    //      accumulating sum / sum-of-squares via LDS float atomics
    const half_t* xb = x + (size_t)blockIdx.x * (2 * 126 * 128);
    for (int item = tid; item < 16 * 252; item += 256) {
        const int q   = item / 252;        // query slot 0..15
        const int rem = item - q * 252;    // rem = c*126 + r
        const int c   = rem / 126;
        const int r   = rem - c * 126;
        v8h chunk = *(const v8h*)(xb + c * (126 * 128) + r * 128 + q * 8);
        *(v8h*)(ldsA + q * KD + rem * 8) = chunk;
        float s1 = 0.f, s2 = 0.f;
#pragma unroll
        for (int e = 0; e < 8; ++e) {
            float v = (float)chunk[e];
            s1 += v;
            s2 += v * v;
        }
        atomicAdd(&ldsXS[q], s1);
        atomicAdd(&ldsX2[q], s2);
    }
    __syncthreads();

    // per-lane query stats: C-tile VGPR r maps to query row r + 8*hi
    float x2s[8], xss[8];
#pragma unroll
    for (int r = 0; r < 8; ++r) {
        x2s[r] = ldsX2[r + 8 * hi];
        xss[r] = ldsXS[r + 8 * hi];
    }

    // ---- running per-lane argmin over this wave's m-tiles
    float bestv[8];
    int   besti[8];
#pragma unroll
    for (int r = 0; r < 8; ++r) { bestv[r] = BIG; besti[r] = 0x7fffffff; }

    const int ntiles32 = (Mrows + 31) >> 5;   // 32-wide macro tiles
    // A fragment source: row = lo, K chunks at +8*hi and +16+8*hi (ISA A layout)
    const half_t* abase = ldsA + lo * KD + hi * 8;

    for (int t = blockIdx.y * WAVES + wave; t < ntiles32; t += NSLOTS) {
        const int m0 = t << 5;
        const int mcol0 = m0 + lo;            // first 16-wide tile column
        const int mcol1 = m0 + 16 + lo;       // second 16-wide tile column
        const int mcl0  = (mcol0 < Mrows) ? mcol0 : (Mrows - 1);
        const int mcl1  = (mcol1 < Mrows) ? mcol1 : (Mrows - 1);
        // B fragment source: column = codebook row; K chunks at +8*hi and
        // +16+8*hi (same order as A)
        const half_t* bbase0 = cb + (size_t)mcl0 * KD + hi * 8;
        const half_t* bbase1 = cb + (size_t)mcl1 * KD + hi * 8;

        v8f acc0 = {};
        v8f acc1 = {};
        for (int k0 = 0; k0 < KD; k0 += 32) {
            union { v16h v; v8h h[2]; } au, bu0, bu1;
            au.h[0]  = *(const v8h*)(abase + k0);        // K = k0 + 8*hi + 0..7
            au.h[1]  = *(const v8h*)(abase + k0 + 16);   // K = k0+16+8*hi + 0..7
            bu0.h[0] = *(const v8h*)(bbase0 + k0);       // K = k0 + 8*hi + 0..7
            bu0.h[1] = *(const v8h*)(bbase0 + k0 + 16);
            bu1.h[0] = *(const v8h*)(bbase1 + k0);
            bu1.h[1] = *(const v8h*)(bbase1 + k0 + 16);
            acc0 = __builtin_amdgcn_wmma_f32_16x16x32_f16(
                false, au.v, false, bu0.v, (short)0, acc0, false, false);
            acc1 = __builtin_amdgcn_wmma_f32_16x16x32_f16(
                false, au.v, false, bu1.v, (short)0, acc1, false, false);
        }

        // ---- distances + running argmin for both 16-wide tiles
        if (mcol0 < Mrows) {
            const float c2v  = c2[mcl0];
            const float cn2v = cn2[mcl0];
            const int i0 = mcol0;
            const int i1 = Mrows + mcol0;
#pragma unroll
            for (int r = 0; r < 8; ++r) {
                const float xcv = acc0[r];
                const float d0v = x2s[r] + c2v  - 2.0f * xcv;
                const float d1v = x2s[r] + cn2v - 2.0f * (xss[r] - xcv);
                if (d0v < bestv[r] || (d0v == bestv[r] && i0 < besti[r])) {
                    bestv[r] = d0v; besti[r] = i0;
                }
                if (d1v < bestv[r] || (d1v == bestv[r] && i1 < besti[r])) {
                    bestv[r] = d1v; besti[r] = i1;
                }
            }
        }
        if (mcol1 < Mrows) {
            const float c2v  = c2[mcl1];
            const float cn2v = cn2[mcl1];
            const int i0 = mcol1;
            const int i1 = Mrows + mcol1;
#pragma unroll
            for (int r = 0; r < 8; ++r) {
                const float xcv = acc1[r];
                const float d0v = x2s[r] + c2v  - 2.0f * xcv;
                const float d1v = x2s[r] + cn2v - 2.0f * (xss[r] - xcv);
                if (d0v < bestv[r] || (d0v == bestv[r] && i0 < besti[r])) {
                    bestv[r] = d0v; besti[r] = i0;
                }
                if (d1v < bestv[r] || (d1v == bestv[r] && i1 < besti[r])) {
                    bestv[r] = d1v; besti[r] = i1;
                }
            }
        }
    }

    // ---- reduce across the 16 lanes that share a query (masks stay in-half)
#pragma unroll
    for (int mask = 1; mask < 16; mask <<= 1) {
#pragma unroll
        for (int r = 0; r < 8; ++r) {
            float ov = __shfl_xor(bestv[r], mask, 32);
            int   oi = __shfl_xor(besti[r], mask, 32);
            if (ov < bestv[r] || (ov == bestv[r] && oi < besti[r])) {
                bestv[r] = ov; besti[r] = oi;
            }
        }
    }

    if (lo == 0) {
        const int slot = blockIdx.y * WAVES + wave;
#pragma unroll
        for (int r = 0; r < 8; ++r) {
            const int q  = r + 8 * hi;                        // query row in tile
            const int Qg = blockIdx.x * 16 + q;               // global query id
            pvals[(size_t)Qg * NSLOTS + slot] = bestv[r];
            pidx [(size_t)Qg * NSLOTS + slot] = besti[r];
        }
    }
}

// ---------------------------------------------------------------------------
// Kernel 3: reduce partial slots per query, expand winning index to 32 bits
// ---------------------------------------------------------------------------
__global__ void finalize_kernel(const float* __restrict__ pvals,
                                const int*   __restrict__ pidx,
                                float* __restrict__ out,
                                int Q_total) {
    const int q = blockIdx.x * blockDim.x + threadIdx.x;
    if (q >= Q_total) return;
    float bv = BIG;
    int   bi = 0x7fffffff;
    for (int s = 0; s < NSLOTS; ++s) {
        float v = pvals[(size_t)q * NSLOTS + s];
        int   i = pidx [(size_t)q * NSLOTS + s];
        if (v < bv || (v == bv && i < bi)) { bv = v; bi = i; }
    }
#pragma unroll
    for (int j = 0; j < 32; ++j) {
        out[(size_t)q * 32 + j] = (float)((bi >> j) & 1);
    }
}

// ---------------------------------------------------------------------------
extern "C" void kernel_launch(void* const* d_in, const int* in_sizes, int n_in,
                              void* d_out, int out_size, void* d_ws, size_t ws_size,
                              hipStream_t stream) {
    const half_t* x  = (const half_t*)d_in[0];   // fp16 [B,2,126,128]
    const half_t* cb = (const half_t*)d_in[1];   // fp16 [M,2,126,8] == [M,2016]
    float* out = (float*)d_out;

    const int Bsz   = in_sizes[0] / (2 * 126 * 128);
    const int Mrows = in_sizes[1] / KD;
    const int Qtot  = Bsz * 16;

    // workspace layout (64B-aligned slabs)
    char* ws = (char*)d_ws;
    size_t off = 0;
    float* c2  = (float*)(ws + off); off += ((size_t)Mrows * 4 + 63) & ~(size_t)63;
    float* cn2 = (float*)(ws + off); off += ((size_t)Mrows * 4 + 63) & ~(size_t)63;
    float* pvals = (float*)(ws + off); off += ((size_t)Qtot * NSLOTS * 4 + 63) & ~(size_t)63;
    int*   pidx  = (int*)  (ws + off); off += ((size_t)Qtot * NSLOTS * 4 + 63) & ~(size_t)63;
    (void)ws_size; (void)n_in; (void)out_size;

    rowstats_kernel<<<Mrows, 256, 0, stream>>>(cb, c2, cn2, Mrows);

    dim3 grid(Bsz, NSPLIT);
    encode_kernel<<<grid, 256, 0, stream>>>(x, cb, c2, cn2, pvals, pidx, Mrows);

    finalize_kernel<<<(Qtot + 255) / 256, 256, 0, stream>>>(pvals, pidx, out, Qtot);
}
